// TrajectoryPredictorCVAE_36275293782620
// MI455X (gfx1250) — matrix operand order repaired and checked by reading, not compiled, not run on hardware
//
#include <hip/hip_runtime.h>
#include <math.h>

// ---------------------------------------------------------------------------
// Types / helpers
// ---------------------------------------------------------------------------
typedef __attribute__((ext_vector_type(16))) __bf16   v16bf;
typedef __attribute__((ext_vector_type(8)))  float    v8f;
typedef __attribute__((ext_vector_type(4)))  unsigned uv4;

union ABFrag { v16bf v; uv4 u[2]; };

__device__ __forceinline__ unsigned short f2bf(float f) {
  unsigned u = __builtin_bit_cast(unsigned, f);
  u += 0x7fffu + ((u >> 16) & 1u);        // round-to-nearest-even
  return (unsigned short)(u >> 16);
}
__device__ __forceinline__ float bf2f(unsigned short h) {
  unsigned u = ((unsigned)h) << 16;
  return __builtin_bit_cast(float, u);
}
__device__ __forceinline__ float sigf(float x) { return 1.f / (1.f + expf(-x)); }

#define BATCH 64
#define TLEN  512
#define NHID  1024
#define NHE   512

// ---------------------------------------------------------------------------
// fp32 -> bf16 convert with zero padding (rows x cols) -> (rowsP x colsP)
// ---------------------------------------------------------------------------
__global__ void k_pad_bf16(const float* __restrict__ src, unsigned short* __restrict__ dst,
                           int rows, int cols, int rowsP, int colsP) {
  int idx = blockIdx.x * 256 + threadIdx.x;
  if (idx >= rowsP * colsP) return;
  int r = idx / colsP, c = idx % colsP;
  float v = (r < rows && c < cols) ? src[(size_t)r * cols + c] : 0.f;
  dst[idx] = f2bf(v);
}

__global__ void k_add_bias(const float* __restrict__ a, const float* __restrict__ b,
                           float* __restrict__ o, int n) {
  int i = blockIdx.x * 256 + threadIdx.x;
  if (i < n) o[i] = a[i] + b[i];
}

// ---------------------------------------------------------------------------
// WMMA GEMM:  C[M, ldc](bf16, at col0) = A[M,K](bf16) @ W[N,K]^T (bf16) + bias
// WG = 256 thr / 8 waves. A tile (64 x 128) staged in LDS and shared by all
// waves; each wave computes 64(M) x 32(N) (2 ntiles: wave, wave+8).
// Out-of-range ntiles (only the tiny N=16 head) compute against clamped row 0
// and are discarded at store time -- keeps the hot loop branch-free so the
// scheduler can pipeline B loads across the unrolled kt iterations.
// Requires K % 128 == 0 (inputs are zero-padded).
// grid.x = ceil((N/16)/16), grid.y = M/64
// ---------------------------------------------------------------------------
__global__ void k_gemm_wmma(const unsigned short* __restrict__ A,
                            const unsigned short* __restrict__ W,
                            const float* __restrict__ bias,
                            unsigned short* __restrict__ C,
                            int M, int N, int K, int ldc, int col0) {
  __shared__ unsigned short ldsA[64 * 128];   // 16 KB

  const int tid  = threadIdx.x;
  const int lane = tid & 31;
  const int wave = tid >> 5;
  const int m0   = blockIdx.y * 64;
  const int half = lane >> 4;
  const int l15  = lane & 15;

  const int nt0 = blockIdx.x * 16 + wave;
  const int nt1 = nt0 + 8;
  const bool has0 = (nt0 * 16) < N;
  const bool has1 = (nt1 * 16) < N;

  const unsigned short* wrow0 =
      W + (size_t)((has0 ? nt0 : 0) * 16 + l15) * K + half * 16;
  const unsigned short* wrow1 =
      W + (size_t)((has1 ? nt1 : 0) * 16 + l15) * K + half * 16;

  v8f acc0[4] = {};
  v8f acc1[4] = {};

  for (int kb = 0; kb < K; kb += 128) {
    __syncthreads();
#pragma unroll
    for (int i = 0; i < 4; ++i) {                   // stage A[m0:m0+64, kb:kb+128]
      int idx = tid + 256 * i;                      // uv4 index (1024 total)
      int row = idx >> 4, cu = idx & 15;
      ((uv4*)ldsA)[idx] = *(const uv4*)(A + (size_t)(m0 + row) * K + kb + cu * 8);
    }
    __syncthreads();
#pragma unroll
    for (int kt = 0; kt < 4; ++kt) {
      int kl = kt * 32;
      __builtin_prefetch(wrow0 + kb + kl + 256, 0, 3);
      __builtin_prefetch(wrow1 + kb + kl + 256, 0, 3);
      ABFrag bf0, bf1;
      bf0.u[0] = *(const uv4*)(wrow0 + kb + kl);
      bf0.u[1] = *(const uv4*)(wrow0 + kb + kl + 8);
      bf1.u[0] = *(const uv4*)(wrow1 + kb + kl);
      bf1.u[1] = *(const uv4*)(wrow1 + kb + kl + 8);
      ABFrag af[4];
#pragma unroll
      for (int mt = 0; mt < 4; ++mt) {
        const unsigned short* ar = ldsA + (16 * mt + l15) * 128 + kl + half * 16;
        af[mt].u[0] = *(const uv4*)(ar);
        af[mt].u[1] = *(const uv4*)(ar + 8);
      }
#pragma unroll
      for (int mt = 0; mt < 4; ++mt)
        acc0[mt] = __builtin_amdgcn_wmma_f32_16x16x32_bf16(
            false, af[mt].v, false, bf0.v, (short)0, acc0[mt], false, false);
#pragma unroll
      for (int mt = 0; mt < 4; ++mt)
        acc1[mt] = __builtin_amdgcn_wmma_f32_16x16x32_bf16(
            false, af[mt].v, false, bf1.v, (short)0, acc1[mt], false, false);
    }
  }

  if (has0) {
    float bv = bias ? bias[nt0 * 16 + l15] : 0.f;
#pragma unroll
    for (int mt = 0; mt < 4; ++mt)
#pragma unroll
      for (int r = 0; r < 8; ++r) {
        int row = m0 + 16 * mt + r + 8 * half;
        C[(size_t)row * ldc + col0 + nt0 * 16 + l15] = f2bf(acc0[mt][r] + bv);
      }
  }
  if (has1) {
    float bv = bias ? bias[nt1 * 16 + l15] : 0.f;
#pragma unroll
    for (int mt = 0; mt < 4; ++mt)
#pragma unroll
      for (int r = 0; r < 8; ++r) {
        int row = m0 + 16 * mt + r + 8 * half;
        C[(size_t)row * ldc + col0 + nt1 * 16 + l15] = f2bf(acc1[mt][r] + bv);
      }
  }
}

// ---------------------------------------------------------------------------
// Grid-wide barrier (atomic counter + generation)
// ---------------------------------------------------------------------------
__device__ __forceinline__ void grid_barrier(volatile unsigned* ctr,
                                             volatile unsigned* gen, unsigned nwg) {
  __syncthreads();
  if (threadIdx.x == 0) {
    __threadfence();
    unsigned g = *gen;
    unsigned old = atomicAdd((unsigned*)ctr, 1u);
    if (old == nwg - 1) {
      *ctr = 0;
      __threadfence();
      atomicAdd((unsigned*)gen, 1u);
    } else {
      while (*gen == g) { __builtin_amdgcn_s_sleep(2); }
      __threadfence();
    }
  }
  __syncthreads();
}

// ---------------------------------------------------------------------------
// Persistent LSTM scan. One WG (256 thr / 8 waves) owns 32 h-lanes:
//   gate rows {g*H + wg*32 .. +32} for g=0..3  (128 rows => 8 ntiles, 1/wave)
// Per step: stage h_prev (64 x 128 K-block) into LDS, WMMA against Whh slice
// streamed from L2, exchange gates through LDS, update c (LDS-resident),
// write h chunk to double-buffered global h, grid barrier.
// xproj = precomputed x@Wih^T + bih + bhh, bf16 [B*T, 4H], row = b*T + t.
// ---------------------------------------------------------------------------
__global__ void k_lstm_scan(const unsigned short* __restrict__ xproj,
                            const unsigned short* __restrict__ Whh,  // [4H, H]
                            unsigned short* __restrict__ hbuf,       // [2][64][H]
                            unsigned short* __restrict__ hs,         // [B*T, ldhs] or null
                            float* __restrict__ hT,                  // [64, H] or null
                            const float* __restrict__ h0, int h0ld,
                            const float* __restrict__ c0, int c0ld,
                            int H, int reverse, int ldhs, int col0,
                            unsigned* sync_ctr, unsigned* sync_gen, int nwg) {
  __shared__ unsigned short lds_h[64 * 128];   // 16 KB A-stage
  __shared__ float gpre[4][32][64];            // 32 KB gate exchange
  __shared__ float cst[32][64];                // 8 KB cell state

  const int tid  = threadIdx.x;
  const int lane = tid & 31;
  const int wave = tid >> 5;
  const int wg   = blockIdx.x;
  const int l15  = lane & 15;
  const int half = lane >> 4;
  const int N4   = 4 * H;
  const int slab = 64 * H;

  // init cell state + our chunk of h (slab 0)
  for (int p = tid; p < 2048; p += 256) {
    int jj = p >> 6, b = p & 63;
    int j = wg * 32 + jj;
    cst[jj][b] = c0 ? c0[b * c0ld + j] : 0.f;
    float hv   = h0 ? h0[b * h0ld + j] : 0.f;
    hbuf[b * H + j] = f2bf(hv);
  }
  grid_barrier(sync_ctr, sync_gen, nwg);

  const int rbase = 16 * wave;         // WG-row base of this wave's ntile
  const int g     = rbase >> 5;        // gate 0..3
  const int jjb   = rbase & 31;        // 0 or 16
  const unsigned short* wrow =
      Whh + (size_t)(g * H + wg * 32 + jjb + l15) * H + half * 16;

  for (int t = 0; t < TLEN; ++t) {
    const int torig = reverse ? (TLEN - 1 - t) : t;
    const unsigned short* hread = hbuf + (t & 1) * slab;
    unsigned short* hwrite      = hbuf + ((t + 1) & 1) * slab;

    v8f acc[4] = {};
    for (int kb = 0; kb < H; kb += 128) {
      __syncthreads();
#pragma unroll
      for (int i = 0; i < 4; ++i) {                 // stage h[:, kb:kb+128]
        int idx = tid + 256 * i;                    // uv4 index (1024 total)
        int row = idx >> 4, cu = idx & 15;
        ((uv4*)lds_h)[idx] = *(const uv4*)(hread + (size_t)row * H + kb + cu * 8);
      }
      __syncthreads();
#pragma unroll
      for (int kt = 0; kt < 4; ++kt) {
        int kl = kt * 32;
        __builtin_prefetch(wrow + kb + kl + 256, 0, 3);
        ABFrag bf;
        bf.u[0] = *(const uv4*)(wrow + kb + kl);
        bf.u[1] = *(const uv4*)(wrow + kb + kl + 8);
#pragma unroll
        for (int mt = 0; mt < 4; ++mt) {
          const unsigned short* ar = lds_h + (16 * mt + l15) * 128 + kl + half * 16;
          ABFrag af;
          af.u[0] = *(const uv4*)(ar);
          af.u[1] = *(const uv4*)(ar + 8);
          acc[mt] = __builtin_amdgcn_wmma_f32_16x16x32_bf16(
              false, af.v, false, bf.v, (short)0, acc[mt], false, false);
        }
      }
    }

    // publish gate pre-activations
#pragma unroll
    for (int mt = 0; mt < 4; ++mt)
#pragma unroll
      for (int r = 0; r < 8; ++r)
        gpre[g][jjb + l15][16 * mt + r + 8 * half] = acc[mt][r];
    __syncthreads();

    // elementwise cell update for our 32 h-lanes x 64 batches
    for (int p = tid; p < 2048; p += 256) {
      int jj = p >> 6, b = p & 63;
      int j = wg * 32 + jj;
      size_t xr = ((size_t)b * TLEN + torig) * N4;
      float gi = gpre[0][jj][b] + bf2f(xproj[xr + j]);
      float gf = gpre[1][jj][b] + bf2f(xproj[xr + H + j]);
      float gg = gpre[2][jj][b] + bf2f(xproj[xr + 2 * H + j]);
      float go = gpre[3][jj][b] + bf2f(xproj[xr + 3 * H + j]);
      float cv = sigf(gf) * cst[jj][b] + sigf(gi) * tanhf(gg);
      float hv = sigf(go) * tanhf(cv);
      cst[jj][b] = cv;
      hwrite[b * H + j] = f2bf(hv);
      if (hs) hs[((size_t)b * TLEN + torig) * ldhs + col0 + j] = f2bf(hv);
      if (hT && t == TLEN - 1) hT[b * H + j] = hv;
    }
    grid_barrier(sync_ctr, sync_gen, nwg);
  }
}

// ---------------------------------------------------------------------------
// Small fp32 FC: out[m,n] = act( bias[n] + sum_k cat(in1,in2,in3)[m,k]*W[n,k] )
// act: 0 none, 1 relu, 2 sigmoid.  in_relu applies relu to in1 on load.
// ---------------------------------------------------------------------------
__global__ void k_fc(const float* __restrict__ in1, int k1, int ld1,
                     const float* __restrict__ in2, int k2, int ld2,
                     const float* __restrict__ in3, int k3, int ld3,
                     const float* __restrict__ W, const float* __restrict__ bias,
                     float* __restrict__ out, int ldo, int M, int N,
                     int act, int in_relu) {
  int idx = blockIdx.x * 256 + threadIdx.x;
  if (idx >= M * N) return;
  int m = idx / N, n = idx % N;
  const float* w = W + (size_t)n * (k1 + k2 + k3);
  float s = bias ? bias[n] : 0.f;
  for (int k = 0; k < k1; ++k) {
    float v = in1[(size_t)m * ld1 + k];
    if (in_relu) v = fmaxf(v, 0.f);
    s += v * w[k];
  }
  w += k1;
  for (int k = 0; k < k2; ++k) s += in2[(size_t)m * ld2 + k] * w[k];
  w += k2;
  for (int k = 0; k < k3; ++k) s += in3[(size_t)m * ld3 + k] * w[k];
  if (act == 1) s = fmaxf(s, 0.f);
  else if (act == 2) s = sigf(s);
  out[(size_t)m * ldo + n] = s;
}

__global__ void k_z(const float* __restrict__ mu, const float* __restrict__ lv,
                    const float* __restrict__ eps, float* __restrict__ z, int n) {
  int i = blockIdx.x * 256 + threadIdx.x;
  if (i >= n) return;
  float l = fminf(fmaxf(lv[i], -10.f), 10.f);
  z[i] = mu[i] + eps[i] * expf(0.5f * l);
}

// din[b*T+t] = [ prev(5) | cond(1024) | pad(123) ]  (bf16, 1152 cols)
__global__ void k_build_din(const float* __restrict__ traj, const float* __restrict__ cond,
                            unsigned short* __restrict__ din) {
  int idx = blockIdx.x * 256 + threadIdx.x;
  if (idx >= 32768 * 1152) return;
  int r = idx / 1152, c = idx % 1152;
  int b = r / TLEN, t = r % TLEN;
  float v;
  if (c < 5)         v = (t == 0) ? 0.f : traj[((size_t)b * TLEN + (t - 1)) * 5 + c];
  else if (c < 1029) v = cond[(size_t)b * NHID + (c - 5)];
  else               v = 0.f;
  din[idx] = f2bf(v);
}

// recon from raw bf16 [32768,16] (+ out bias) -> d_out[0 .. 163839] f32
__global__ void k_recon(const unsigned short* __restrict__ raw,
                        const float* __restrict__ ob, float* __restrict__ out) {
  int idx = blockIdx.x * 256 + threadIdx.x;
  if (idx >= 32768 * 5) return;
  int row = idx / 5, c = idx % 5;
  float x = bf2f(raw[(size_t)row * 16 + c]) + ob[c];
  float v;
  if (c == 2)      v = (x > 20.f) ? x : log1pf(expf(x));  // softplus
  else if (c == 4) v = sigf(x);
  else             v = tanhf(x);
  out[idx] = v;
}

// ---------------------------------------------------------------------------
// Host orchestration
// ---------------------------------------------------------------------------
enum {
  IN_CTX = 0, IN_TRAJ = 1, IN_EPS = 2,
  DEC_FC1_W = 3, DEC_FC1_B = 4, DEC_FC2_W = 5, DEC_FC2_B = 6,
  DL0_WHH = 7, DL0_WIH = 8, DL0_BHH = 9, DL0_BIH = 10,
  DL1_WHH = 11, DL1_WIH = 12, DL1_BHH = 13, DL1_BIH = 14,
  ENC_FC_W = 15, ENC_FC_B = 16,
  E0B_WHH = 17, E0B_WIH = 18, E0B_BHH = 19, E0B_BIH = 20,
  E0F_WHH = 21, E0F_WIH = 22, E0F_BHH = 23, E0F_BIH = 24,
  E1B_WHH = 25, E1B_WIH = 26, E1B_BHH = 27, E1B_BIH = 28,
  E1F_WHH = 29, E1F_WIH = 30, E1F_BHH = 31, E1F_BIH = 32,
  LOGVAR_W = 33, LOGVAR_B = 34, MU_W = 35, MU_B = 36,
  LEN1_W = 37, LEN1_B = 38, LEN2_W = 39, LEN2_B = 40,
  OUT_W = 41, OUT_B = 42
};

static inline dim3 g1(size_t n) { return dim3((unsigned)((n + 255) / 256)); }

extern "C" void kernel_launch(void* const* d_in, const int* in_sizes, int n_in,
                              void* d_out, int out_size, void* d_ws, size_t ws_size,
                              hipStream_t stream) {
  const int MT = BATCH * TLEN;  // 32768
  size_t off = 0;
  auto alloc = [&](size_t bytes) -> void* {
    void* p = (char*)d_ws + off;
    off += (bytes + 255) & ~(size_t)255;
    return p;
  };

  unsigned short* xproj = (unsigned short*)alloc((size_t)MT * 4096 * 2);
  unsigned short* x_pad = (unsigned short*)alloc((size_t)MT * 128 * 2);
  unsigned short* x2    = (unsigned short*)alloc((size_t)MT * 1024 * 2);
  unsigned short* din   = (unsigned short*)alloc((size_t)MT * 1152 * 2);
  unsigned short* hs0   = (unsigned short*)alloc((size_t)MT * 1024 * 2);
  unsigned short* hs1   = (unsigned short*)alloc((size_t)MT * 1024 * 2);
  unsigned short* raw   = (unsigned short*)alloc((size_t)MT * 16 * 2);
  unsigned short* hbuf  = (unsigned short*)alloc((size_t)2 * 64 * 1024 * 2);

  unsigned short* WihF1 = (unsigned short*)alloc((size_t)2048 * 128 * 2);
  unsigned short* WihB1 = (unsigned short*)alloc((size_t)2048 * 128 * 2);
  unsigned short* WhhF1 = (unsigned short*)alloc((size_t)2048 * 512 * 2);
  unsigned short* WhhB1 = (unsigned short*)alloc((size_t)2048 * 512 * 2);
  unsigned short* WihF2 = (unsigned short*)alloc((size_t)2048 * 1024 * 2);
  unsigned short* WihB2 = (unsigned short*)alloc((size_t)2048 * 1024 * 2);
  unsigned short* WhhF2 = (unsigned short*)alloc((size_t)2048 * 512 * 2);
  unsigned short* WhhB2 = (unsigned short*)alloc((size_t)2048 * 512 * 2);
  unsigned short* WihD0 = (unsigned short*)alloc((size_t)4096 * 1152 * 2);
  unsigned short* WhhD0 = (unsigned short*)alloc((size_t)4096 * 1024 * 2);
  unsigned short* WihD1 = (unsigned short*)alloc((size_t)4096 * 1024 * 2);
  unsigned short* WhhD1 = (unsigned short*)alloc((size_t)4096 * 1024 * 2);
  unsigned short* Wout  = (unsigned short*)alloc((size_t)16 * 1024 * 2);

  float* bF1 = (float*)alloc(2048 * 4);
  float* bB1 = (float*)alloc(2048 * 4);
  float* bF2 = (float*)alloc(2048 * 4);
  float* bB2 = (float*)alloc(2048 * 4);
  float* bD0 = (float*)alloc(4096 * 4);
  float* bD1 = (float*)alloc(4096 * 4);

  float* hTf2    = (float*)alloc((size_t)64 * 512 * 4);
  float* hTb2    = (float*)alloc((size_t)64 * 512 * 4);
  float* henc    = (float*)alloc((size_t)64 * 1024 * 4);
  float* zbuf    = (float*)alloc((size_t)64 * 128 * 4);
  float* cond    = (float*)alloc((size_t)64 * 1024 * 4);
  float* decinit = (float*)alloc((size_t)64 * 2048 * 4);
  float* lenh    = (float*)alloc((size_t)64 * 512 * 4);
  unsigned* sync = (unsigned*)alloc(256);

  hipMemsetAsync(sync, 0, 8, stream);

  const float* ctx  = (const float*)d_in[IN_CTX];
  const float* traj = (const float*)d_in[IN_TRAJ];
  const float* eps  = (const float*)d_in[IN_EPS];
  float* out_recon  = (float*)d_out;                 // 163840
  float* out_mu     = (float*)d_out + 163840;        // 8192
  float* out_lv     = (float*)d_out + 172032;        // 8192
  float* out_ratio  = (float*)d_out + 180224;        // 64

  auto pad = [&](int src, unsigned short* dst, int r, int c, int rP, int cP) {
    k_pad_bf16<<<g1((size_t)rP * cP), 256, 0, stream>>>(
        (const float*)d_in[src], dst, r, c, rP, cP);
  };
  // weights -> bf16 (K padded to mult. of 128)
  pad(E0F_WIH, WihF1, 2048, 5, 2048, 128);
  pad(E0B_WIH, WihB1, 2048, 5, 2048, 128);
  pad(E0F_WHH, WhhF1, 2048, 512, 2048, 512);
  pad(E0B_WHH, WhhB1, 2048, 512, 2048, 512);
  pad(E1F_WIH, WihF2, 2048, 1024, 2048, 1024);
  pad(E1B_WIH, WihB2, 2048, 1024, 2048, 1024);
  pad(E1F_WHH, WhhF2, 2048, 512, 2048, 512);
  pad(E1B_WHH, WhhB2, 2048, 512, 2048, 512);
  pad(DL0_WIH, WihD0, 4096, 1029, 4096, 1152);
  pad(DL0_WHH, WhhD0, 4096, 1024, 4096, 1024);
  pad(DL1_WIH, WihD1, 4096, 1024, 4096, 1024);
  pad(DL1_WHH, WhhD1, 4096, 1024, 4096, 1024);
  pad(OUT_W,   Wout,  5, 1024, 16, 1024);
  // x -> bf16 padded [32768, 128]
  k_pad_bf16<<<g1((size_t)MT * 128), 256, 0, stream>>>(traj, x_pad, MT, 5, MT, 128);

  auto bias2 = [&](int bih, int bhh, float* dst, int n) {
    k_add_bias<<<g1(n), 256, 0, stream>>>((const float*)d_in[bih],
                                          (const float*)d_in[bhh], dst, n);
  };
  bias2(E0F_BIH, E0F_BHH, bF1, 2048);
  bias2(E0B_BIH, E0B_BHH, bB1, 2048);
  bias2(E1F_BIH, E1F_BHH, bF2, 2048);
  bias2(E1B_BIH, E1B_BHH, bB2, 2048);
  bias2(DL0_BIH, DL0_BHH, bD0, 4096);
  bias2(DL1_BIH, DL1_BHH, bD1, 4096);

  auto gemm = [&](const unsigned short* A, const unsigned short* W, const float* b,
                  unsigned short* C, int N, int K, int ldc, int col0) {
    dim3 grid((N / 16 + 15) / 16, MT / 64);
    k_gemm_wmma<<<grid, 256, 0, stream>>>(A, W, b, C, MT, N, K, ldc, col0);
  };
  auto scan = [&](const unsigned short* xp, const unsigned short* Whh,
                  unsigned short* hs, float* hT, const float* h0, int h0ld,
                  const float* c0, int c0ld, int H, int rev, int ldhs, int col0) {
    int nwg = H / 32;
    k_lstm_scan<<<dim3(nwg), 256, 0, stream>>>(xp, Whh, hbuf, hs, hT, h0, h0ld,
                                               c0, c0ld, H, rev, ldhs, col0,
                                               sync, sync + 1, nwg);
  };

  // ---- encoder layer 1 (H=512), hs -> x2 halves ----
  gemm(x_pad, WihF1, bF1, xproj, 2048, 128, 2048, 0);
  scan(xproj, WhhF1, x2, nullptr, nullptr, 0, nullptr, 0, 512, 0, 1024, 0);
  gemm(x_pad, WihB1, bB1, xproj, 2048, 128, 2048, 0);
  scan(xproj, WhhB1, x2, nullptr, nullptr, 0, nullptr, 0, 512, 1, 1024, 512);

  // ---- encoder layer 2 (H=512), only hT needed ----
  gemm(x2, WihF2, bF2, xproj, 2048, 1024, 2048, 0);
  scan(xproj, WhhF2, nullptr, hTf2, nullptr, 0, nullptr, 0, 512, 0, 0, 0);
  gemm(x2, WihB2, bB2, xproj, 2048, 1024, 2048, 0);
  scan(xproj, WhhB2, nullptr, hTb2, nullptr, 0, nullptr, 0, 512, 1, 0, 0);

  // ---- bottleneck (fp32 small FCs) ----
  k_fc<<<g1(64 * 1024), 256, 0, stream>>>(hTf2, 512, 512, hTb2, 512, 512, ctx, 5, 5,
      (const float*)d_in[ENC_FC_W], (const float*)d_in[ENC_FC_B],
      henc, 1024, 64, 1024, 1, 0);
  k_fc<<<g1(64 * 128), 256, 0, stream>>>(henc, 1024, 1024, nullptr, 0, 0, nullptr, 0, 0,
      (const float*)d_in[MU_W], (const float*)d_in[MU_B], out_mu, 128, 64, 128, 0, 0);
  k_fc<<<g1(64 * 128), 256, 0, stream>>>(henc, 1024, 1024, nullptr, 0, 0, nullptr, 0, 0,
      (const float*)d_in[LOGVAR_W], (const float*)d_in[LOGVAR_B], out_lv, 128, 64, 128, 0, 0);
  k_z<<<g1(64 * 128), 256, 0, stream>>>(out_mu, out_lv, eps, zbuf, 64 * 128);
  k_fc<<<g1(64 * 1024), 256, 0, stream>>>(zbuf, 128, 128, ctx, 5, 5, nullptr, 0, 0,
      (const float*)d_in[DEC_FC1_W], (const float*)d_in[DEC_FC1_B], cond, 1024, 64, 1024, 0, 0);
  k_fc<<<g1(64 * 2048), 256, 0, stream>>>(cond, 1024, 1024, nullptr, 0, 0, nullptr, 0, 0,
      (const float*)d_in[DEC_FC2_W], (const float*)d_in[DEC_FC2_B],
      decinit, 2048, 64, 2048, 0, 1);

  // ---- decoder ----
  k_build_din<<<g1((size_t)MT * 1152), 256, 0, stream>>>(traj, cond, din);
  gemm(din, WihD0, bD0, xproj, 4096, 1152, 4096, 0);
  scan(xproj, WhhD0, hs0, nullptr, decinit, 2048, decinit + 1024, 2048, 1024, 0, 1024, 0);
  gemm(hs0, WihD1, bD1, xproj, 4096, 1024, 4096, 0);
  scan(xproj, WhhD1, hs1, nullptr, decinit, 2048, decinit + 1024, 2048, 1024, 0, 1024, 0);

  // ---- output head ----
  gemm(hs1, Wout, nullptr, raw, 16, 1024, 16, 0);
  k_recon<<<g1((size_t)MT * 5), 256, 0, stream>>>(raw, (const float*)d_in[OUT_B], out_recon);

  // ---- length ratio head ----
  k_fc<<<g1(64 * 512), 256, 0, stream>>>(ctx, 5, 5, nullptr, 0, 0, nullptr, 0, 0,
      (const float*)d_in[LEN1_W], (const float*)d_in[LEN1_B], lenh, 512, 64, 512, 1, 0);
  k_fc<<<g1(64), 256, 0, stream>>>(lenh, 512, 512, nullptr, 0, 0, nullptr, 0, 0,
      (const float*)d_in[LEN2_W], (const float*)d_in[LEN2_B], out_ratio, 1, 64, 1, 2, 0);

  (void)in_sizes; (void)n_in; (void)out_size; (void)ws_size;
}
